// LASCC_41858751267181
// MI455X (gfx1250) — matrix operand order repaired and checked by reading, not compiled, not run on hardware
//
#include <hip/hip_runtime.h>
#include <math.h>

// ---------------------------------------------------------------------------
// Sizes (fixed by the reference): b=16, c=128, h=w=64, ps=2
//   nph=npw=32, npatch=1024, K=128 channels, q in {0,1,2} (p0, p1, avg)
// ---------------------------------------------------------------------------
#define NB       16
#define NC       128
#define NPATCH   1024
#define NK       128
#define TWO_SR2  5.12f          // 2 * (32*0.05)^2
#define WAVES    4              // waves per block in the main kernels

typedef __attribute__((ext_vector_type(16))) _Float16 v16h;
typedef __attribute__((ext_vector_type(8)))  float    v8f;
typedef __attribute__((ext_vector_type(4)))  float    f32x4;

// workspace layout (bytes)
#define XH_BYTES   ((size_t)NB * 2 * NPATCH * NK * 2)        // 8,388,608  f16
#define S_BYTES    ((size_t)NB * 3 * NPATCH * 4)             //   196,608  f32

__device__ inline v8f zero8() {
  v8f z = {0.f,0.f,0.f,0.f,0.f,0.f,0.f,0.f};
  return z;
}

// ---------------------------------------------------------------------------
// A fragment: rows rowBase..+15, K chunk kk..kk+31 of row-major [n][128] f16.
// ISA 16-bit A 16x32 layout: lane<16 -> K {kk..kk+7, kk+16..kk+23},
//                            lane>=16 -> K {kk+8..kk+15, kk+24..kk+31}.
// ---------------------------------------------------------------------------
__device__ inline v16h load_a_frag(const _Float16* base, int rowBase, int kk, int lane) {
  const _Float16* p = base + (size_t)(rowBase + (lane & 15)) * NK + kk + ((lane >> 4) << 3);
  v16h a;
  f32x4* ap = reinterpret_cast<f32x4*>(&a);
  ap[0] = *reinterpret_cast<const f32x4*>(p);        // 8 halves
  ap[1] = *reinterpret_cast<const f32x4*>(p + 16);   // 8 halves
  return a;
}

// B fragment read from an LDS-staged tile laid out [col 0..15][k 0..127].
// lane<16 -> K {kk..kk+7, kk+16..kk+23} of col (lane), lane>=16 -> K {kk+8..kk+15, kk+24..kk+31}.
__device__ inline v16h load_b_frag_lds(const _Float16* buf, int kk, int lane) {
  const _Float16* p = buf + (lane & 15) * NK + kk + ((lane >> 4) << 3);
  v16h b;
  f32x4* bp = reinterpret_cast<f32x4*>(&b);
  bp[0] = *reinterpret_cast<const f32x4*>(p);
  bp[1] = *reinterpret_cast<const f32x4*>(p + 16);
  return b;
}

// ---------------------------------------------------------------------------
// Stage one column tile (16 cols x 128 K, both matrices = 8KB, contiguous in
// memory) into this wave's LDS buffer with CDNA5 async global->LDS copies.
// Each GLOBAL_LOAD_ASYNC_TO_LDS_B128 moves 32 lanes x 16B = 512B; the 24-bit
// instruction offset advances BOTH the LDS and global addresses.
// ---------------------------------------------------------------------------
__device__ inline void stage_tile_async(_Float16* dst, const _Float16* s0,
                                        const _Float16* s1, int lane) {
  // close WAR hazard: previous tile's ds_load_b128 B-fragment reads must
  // retire before the async engine overwrites the buffer
  asm volatile("s_wait_dscnt 0" ::: "memory");
  unsigned d0 = (unsigned)(size_t)dst + (unsigned)(lane << 4);   // LDS offset
  unsigned d1 = d0 + 4096u;
  unsigned long long a0 = (unsigned long long)(size_t)s0 + (unsigned)(lane << 4);
  unsigned long long a1 = (unsigned long long)(size_t)s1 + (unsigned)(lane << 4);
  asm volatile(
      "global_load_async_to_lds_b128 %0, %2, off\n\t"
      "global_load_async_to_lds_b128 %0, %2, off offset:512\n\t"
      "global_load_async_to_lds_b128 %0, %2, off offset:1024\n\t"
      "global_load_async_to_lds_b128 %0, %2, off offset:1536\n\t"
      "global_load_async_to_lds_b128 %0, %2, off offset:2048\n\t"
      "global_load_async_to_lds_b128 %0, %2, off offset:2560\n\t"
      "global_load_async_to_lds_b128 %0, %2, off offset:3072\n\t"
      "global_load_async_to_lds_b128 %0, %2, off offset:3584\n\t"
      "global_load_async_to_lds_b128 %1, %3, off\n\t"
      "global_load_async_to_lds_b128 %1, %3, off offset:512\n\t"
      "global_load_async_to_lds_b128 %1, %3, off offset:1024\n\t"
      "global_load_async_to_lds_b128 %1, %3, off offset:1536\n\t"
      "global_load_async_to_lds_b128 %1, %3, off offset:2048\n\t"
      "global_load_async_to_lds_b128 %1, %3, off offset:2560\n\t"
      "global_load_async_to_lds_b128 %1, %3, off offset:3072\n\t"
      "global_load_async_to_lds_b128 %1, %3, off offset:3584"
      :: "v"(d0), "v"(d1), "v"(a0), "v"(a1) : "memory");
}

__device__ inline void wait_async_done() {
  asm volatile("s_wait_asynccnt 0" ::: "memory");
}

// 16x16 tile of X*X^T over K=128 for both matrices; A from registers, B from LDS.
__device__ inline void corr_tile_lds(const v16h* a0, const v16h* a1,
                                     const _Float16* bufB, int lane,
                                     v8f& c0, v8f& c1) {
#pragma unroll
  for (int k4 = 0; k4 < 4; ++k4) {
    v16h b0 = load_b_frag_lds(bufB,            k4 * 32, lane);
    v16h b1 = load_b_frag_lds(bufB + 16 * NK,  k4 * 32, lane);
    c0 = __builtin_amdgcn_wmma_f32_16x16x32_f16(false, a0[k4], false, b0,
                                                (short)0, c0, false, false);
    c1 = __builtin_amdgcn_wmma_f32_16x16x32_f16(false, a1[k4], false, b1,
                                                (short)0, c1, false, false);
  }
}

// ---------------------------------------------------------------------------
// Kernel 0: L2-normalize channel vectors at the two diagonal pixels of each
// 2x2 patch, write f16 matrix Xh[b][p][n][c]  (row-major, K contiguous).
// ---------------------------------------------------------------------------
__global__ void lascc_prep(const float* __restrict__ x, _Float16* __restrict__ Xh) {
  int bpn = blockIdx.x;              // b*2048 + p*1024 + n
  int n = bpn & (NPATCH - 1);
  int p = (bpn >> 10) & 1;
  int b = bpn >> 11;
  int c = threadIdx.x;
  int h = ((n >> 5) << 1) + p;
  int w = ((n & 31) << 1) + p;
  float v = x[(((size_t)b * NC + c) * 64 + h) * 64 + w];

  float ss = v * v;
#pragma unroll
  for (int off = 16; off > 0; off >>= 1)
    ss += __shfl_down(ss, off, 32);
  __shared__ float red[4];
  if ((threadIdx.x & 31) == 0) red[threadIdx.x >> 5] = ss;
  __syncthreads();
  float tot = red[0] + red[1] + red[2] + red[3];
  float inv = 1.0f / fmaxf(sqrtf(tot), 1e-12f);
  Xh[(size_t)bpn * NK + c] = (_Float16)(v * inv);
}

// ---------------------------------------------------------------------------
// Kernel 1: row sums  S[b][q][n] = sum_m exp(alpha*(1-g[n,m])*corr_q[n,m]).
// grid (64 row-blocks, 16 batches) x 4 waves; wave w owns column tiles
// w*16..w*16+15. B staged via async LDS copies; per-row sums reduced with
// shfl_xor inside each 16-lane half, merged across waves through LDS.
// ---------------------------------------------------------------------------
__global__ void __launch_bounds__(WAVES * 32)
lascc_rowsum(const _Float16* __restrict__ Xh,
             const float* __restrict__ alphaPtr,
             float* __restrict__ S) {
  __shared__ _Float16 Bbuf[WAVES][2 * 16 * NK];   // 8KB per wave
  __shared__ float g1[32];
  __shared__ float sred[WAVES][3][16];

  int tid = threadIdx.x;
  int wave = tid >> 5, lane = tid & 31;
  if (tid < 32) g1[tid] = __expf(-(float)(tid * tid) * (1.0f / TWO_SR2));

  int rowBase = blockIdx.x << 4;
  int b = blockIdx.y;
  float alpha = alphaPtr[0];
  const _Float16* X0 = Xh + (size_t)b * 2 * NPATCH * NK;
  const _Float16* X1 = X0 + (size_t)NPATCH * NK;

  // A fragments for this row block, both matrices (held in 64 VGPRs)
  v16h a0[4], a1[4];
#pragma unroll
  for (int k4 = 0; k4 < 4; ++k4) {
    a0[k4] = load_a_frag(X0, rowBase, k4 * 32, lane);
    a1[k4] = load_a_frag(X1, rowBase, k4 * 32, lane);
  }

  float rs[3][8];                 // running row sums (replicated per half)
#pragma unroll
  for (int q = 0; q < 3; ++q)
#pragma unroll
    for (int e = 0; e < 8; ++e) rs[q][e] = 0.f;

  int hi = lane >> 4;
  __syncthreads();                // g1 ready

  for (int i = 0; i < 16; ++i) {
    int ct = (wave << 4) + i;
    int colBase = ct << 4;
    stage_tile_async(Bbuf[wave], X0 + (size_t)colBase * NK,
                     X1 + (size_t)colBase * NK, lane);
    wait_async_done();

    v8f c0 = zero8(), c1 = zero8();
    corr_tile_lds(a0, a1, Bbuf[wave], lane, c0, c1);

    int col = colBase + (lane & 15);
    int cr = col >> 5, cc = col & 31;
#pragma unroll
    for (int e = 0; e < 8; ++e) {
      int row = rowBase + e + (hi << 3);
      int rr = row >> 5, rc = row & 31;
      int dr = (rr < cr) ? cr - rr : rr - cr;
      int dc = (rc < cc) ? cc - rc : rc - cc;
      float am = alpha * (1.0f - g1[dr] * g1[dc]);
      float e0 = __expf(am * c0[e]);
      float e1 = __expf(am * c1[e]);
      float e2 = __expf(am * 0.5f * (c0[e] + c1[e]));
      // sum across the 16 lanes of this half (columns of the tile)
#pragma unroll
      for (int m = 1; m < 16; m <<= 1) {
        e0 += __shfl_xor(e0, m, 32);
        e1 += __shfl_xor(e1, m, 32);
        e2 += __shfl_xor(e2, m, 32);
      }
      rs[0][e] += e0; rs[1][e] += e1; rs[2][e] += e2;
    }
  }

  if ((lane & 15) == 0) {
#pragma unroll
    for (int q = 0; q < 3; ++q)
#pragma unroll
      for (int e = 0; e < 8; ++e)
        sred[wave][q][(hi << 3) + e] = rs[q][e];
  }
  __syncthreads();
  if (tid < 48) {
    int q = tid >> 4, r = tid & 15;
    float s = sred[0][q][r] + sred[1][q][r] + sred[2][q][r] + sred[3][q][r];
    S[((size_t)b * 3 + q) * NPATCH + rowBase + r] = s;
  }
}

// ---------------------------------------------------------------------------
// Kernel 2: a_q[n,m] = exp(s)^2/(S[n]*S[m]) (s symmetric => rowSM*colSM),
// per-row top-3 over the column tiles; per-wave partial top-3 merged in LDS.
// ---------------------------------------------------------------------------
__global__ void __launch_bounds__(WAVES * 32)
lascc_topk(const _Float16* __restrict__ Xh,
           const float* __restrict__ alphaPtr,
           const float* __restrict__ S,
           float* __restrict__ Top3) {
  __shared__ _Float16 Bbuf[WAVES][2 * 16 * NK];   // 8KB per wave
  __shared__ float tl[WAVES][3][256];
  __shared__ float g1[32];
  __shared__ float rowInvL[3][16];
  __shared__ float tred[WAVES][3][16][3];

  int tid = threadIdx.x;
  int wave = tid >> 5, lane = tid & 31;
  if (tid < 32) g1[tid] = __expf(-(float)(tid * tid) * (1.0f / TWO_SR2));

  int rowBase = blockIdx.x << 4;
  int b = blockIdx.y;
  float alpha = alphaPtr[0];
  const _Float16* X0 = Xh + (size_t)b * 2 * NPATCH * NK;
  const _Float16* X1 = X0 + (size_t)NPATCH * NK;

  if (tid < 48) {
    int q = tid >> 4, r = tid & 15;
    rowInvL[q][r] = 1.0f / S[((size_t)b * 3 + q) * NPATCH + rowBase + r];
  }

  v16h a0[4], a1[4];
#pragma unroll
  for (int k4 = 0; k4 < 4; ++k4) {
    a0[k4] = load_a_frag(X0, rowBase, k4 * 32, lane);
    a1[k4] = load_a_frag(X1, rowBase, k4 * 32, lane);
  }

  float t0[3], t1[3], t2[3];      // per-wave running top-3 (lanes 0..15, row=lane)
#pragma unroll
  for (int q = 0; q < 3; ++q) { t0[q] = -3.4e38f; t1[q] = -3.4e38f; t2[q] = -3.4e38f; }

  int hi = lane >> 4;
  __syncthreads();                // g1 / rowInvL ready

  for (int i = 0; i < 16; ++i) {
    int ct = (wave << 4) + i;
    int colBase = ct << 4;
    stage_tile_async(Bbuf[wave], X0 + (size_t)colBase * NK,
                     X1 + (size_t)colBase * NK, lane);
    wait_async_done();

    v8f c0 = zero8(), c1 = zero8();
    corr_tile_lds(a0, a1, Bbuf[wave], lane, c0, c1);

    int col = colBase + (lane & 15);
    int cr = col >> 5, cc = col & 31;
    float ci0 = 1.0f / S[((size_t)b * 3 + 0) * NPATCH + col];
    float ci1 = 1.0f / S[((size_t)b * 3 + 1) * NPATCH + col];
    float ci2 = 1.0f / S[((size_t)b * 3 + 2) * NPATCH + col];
#pragma unroll
    for (int e = 0; e < 8; ++e) {
      int rl = e + (hi << 3);
      int row = rowBase + rl;
      int rr = row >> 5, rc = row & 31;
      int dr = (rr < cr) ? cr - rr : rr - cr;
      int dc = (rc < cc) ? cc - rc : rc - cc;
      float am = alpha * (1.0f - g1[dr] * g1[dc]);
      int idx = rl * 16 + (lane & 15);
      tl[wave][0][idx] = __expf(2.0f * am * c0[e]) * rowInvL[0][rl] * ci0;
      tl[wave][1][idx] = __expf(2.0f * am * c1[e]) * rowInvL[1][rl] * ci1;
      tl[wave][2][idx] = __expf(am * (c0[e] + c1[e])) * rowInvL[2][rl] * ci2;
    }
    asm volatile("s_wait_dscnt 0" ::: "memory");   // in-wave: writes visible
    if (lane < 16) {
#pragma unroll
      for (int q = 0; q < 3; ++q) {
#pragma unroll
        for (int j = 0; j < 16; ++j) {
          float v = tl[wave][q][lane * 16 + j];
          if (v > t0[q])      { t2[q] = t1[q]; t1[q] = t0[q]; t0[q] = v; }
          else if (v > t1[q]) { t2[q] = t1[q]; t1[q] = v; }
          else if (v > t2[q]) { t2[q] = v; }
        }
      }
    }
  }

  if (lane < 16) {
#pragma unroll
    for (int q = 0; q < 3; ++q) {
      tred[wave][q][lane][0] = t0[q];
      tred[wave][q][lane][1] = t1[q];
      tred[wave][q][lane][2] = t2[q];
    }
  }
  __syncthreads();
  if (tid < 48) {
    int q = tid >> 4, r = tid & 15;
    float m0 = -3.4e38f, m1 = -3.4e38f, m2 = -3.4e38f;
#pragma unroll
    for (int w = 0; w < WAVES; ++w)
#pragma unroll
      for (int k = 0; k < 3; ++k) {
        float v = tred[w][q][r][k];
        if (v > m0)      { m2 = m1; m1 = m0; m0 = v; }
        else if (v > m1) { m2 = m1; m1 = v; }
        else if (v > m2) { m2 = v; }
      }
    float* o = Top3 + (((size_t)b * 3 + q) * NPATCH + rowBase + r) * 3;
    o[0] = m0; o[1] = m1; o[2] = m2;
  }
}

// ---------------------------------------------------------------------------
// Kernel 3: scatter  out[b][k][h][w] = Top3[b][q(h%2,w%2)][n(h/2,w/2)][k]
// with src = [0,2,2,1]  ->  q = 0 for pp==0, 1 for pp==3, 2 otherwise.
// ---------------------------------------------------------------------------
__global__ void lascc_scatter(const float* __restrict__ Top3, float* __restrict__ out) {
  int idx = blockIdx.x * 256 + threadIdx.x;
  if (idx >= NB * 3 * 64 * 64) return;
  int w = idx & 63;
  int h = (idx >> 6) & 63;
  int bk = idx >> 12;
  int k = bk % 3;
  int b = bk / 3;
  int n = ((h >> 1) << 5) + (w >> 1);
  int pp = ((h & 1) << 1) + (w & 1);
  int q = (pp == 0) ? 0 : ((pp == 3) ? 1 : 2);
  out[idx] = Top3[(((size_t)b * 3 + q) * NPATCH + n) * 3 + k];
}

// ---------------------------------------------------------------------------
extern "C" void kernel_launch(void* const* d_in, const int* in_sizes, int n_in,
                              void* d_out, int out_size, void* d_ws, size_t ws_size,
                              hipStream_t stream) {
  (void)in_sizes; (void)n_in; (void)out_size; (void)ws_size;
  const float* x     = (const float*)d_in[0];
  const float* alpha = (const float*)d_in[1];
  float* out = (float*)d_out;

  char* ws = (char*)d_ws;
  _Float16* Xh  = (_Float16*)ws;
  float*    S   = (float*)(ws + XH_BYTES);
  float*    T3  = (float*)(ws + XH_BYTES + S_BYTES);

  lascc_prep<<<NB * 2 * NPATCH, NC, 0, stream>>>(x, Xh);
  lascc_rowsum<<<dim3(NPATCH / 16, NB), WAVES * 32, 0, stream>>>(Xh, alpha, S);
  lascc_topk<<<dim3(NPATCH / 16, NB), WAVES * 32, 0, stream>>>(Xh, alpha, S, T3);
  lascc_scatter<<<(NB * 3 * 64 * 64 + 255) / 256, 256, 0, stream>>>(T3, out);
}
